// HebbianNetwork_22153441313279
// MI455X (gfx1250) — compile-verified, matrix operand order checked by reading, not run on hardware
//
#include <hip/hip_runtime.h>
#include <hip/hip_bf16.h>

#define ISIZE 512
#define REC   1024
#define NACT  18
#define BS    64
#define CLIPVAL 2.0f

typedef __attribute__((ext_vector_type(16))) _Float16 v16h;
typedef __attribute__((ext_vector_type(8)))  float    v8f;

// ---------------------------------------------------------------------------
// Kernel 1: partial[c][b][j] = sum_{i in chunk c} rv[b,i]*(w[i,j]+alpha[i,j]*hebb[b,i,j])
// grid = BS*4 blocks, 256 threads; each thread owns 4 consecutive j (float4).
// Streams hebb exactly once (268 MB) — bandwidth bound; unroll 4 for MLP.
// ---------------------------------------------------------------------------
__global__ void plastic_reduce_kernel(const float* __restrict__ rv,
                                      const float* __restrict__ hebb,
                                      const float* __restrict__ w,
                                      const float* __restrict__ alpha,
                                      float* __restrict__ partial) {
    const int b = blockIdx.x >> 2;
    const int c = blockIdx.x & 3;
    const int j0 = threadIdx.x * 4;

    const float* rvb = rv + b * REC;
    const size_t hebb_b = (size_t)b * REC * REC;

    float ax = 0.f, ay = 0.f, az = 0.f, aw = 0.f;

    const int i_begin = c * (REC / 4);
    const int i_end   = i_begin + (REC / 4);
#pragma unroll 4
    for (int i = i_begin; i < i_end; ++i) {
        const float s = rvb[i];
        const float4 w4 = *(const float4*)(w     + (size_t)i * REC + j0);
        const float4 a4 = *(const float4*)(alpha + (size_t)i * REC + j0);
        const float4 h4 = *(const float4*)(hebb + hebb_b + (size_t)i * REC + j0);
        ax += s * (w4.x + a4.x * h4.x);
        ay += s * (w4.y + a4.y * h4.y);
        az += s * (w4.z + a4.z * h4.z);
        aw += s * (w4.w + a4.w * h4.w);
    }
    float4 out;
    out.x = ax; out.y = ay; out.z = az; out.w = aw;
    *(float4*)(partial + ((size_t)c * BS + b) * REC + j0) = out;
}

// ---------------------------------------------------------------------------
// Kernel 2: hactiv = tanh(X @ W0^T + b0 + rec_in) via v_wmma_f32_16x16x32_f16.
// One wave per 16x16 output tile: 4 M-tiles (batch) x 64 N-tiles = 256 waves.
// Launch: 64 blocks x 128 threads (4 waves/block). EXEC all-ones, no divergence.
// K-loop unrolled 4x: WMMA (tracked as TRANS) co-executes with the
// v_cvt_pk_f16_f32 staging of the next step.
//
// A (16x32 f16): lane L<16 -> row M=L, VGPR0..3 hold K pairs {0..7}, VGPR4..7
// hold K {16..23}; lanes 16..31 shift K by +8. So per lane the 16 halves are
// the 8 floats at k0+8*khalf and the 8 floats at k0+16+8*khalf. B (32x16) is
// symmetric with N=lane%16, reading rows of W0 (since B[k][n] = W0[n][k]).
// C/D (16x16 f32, 8 VGPRs): M = r + 8*(lane/16), N = lane%16.
// ---------------------------------------------------------------------------
__global__ void wmma_hactiv_kernel(const float* __restrict__ X,
                                   const float* __restrict__ W0,
                                   const float* __restrict__ b0,
                                   const float* __restrict__ partial,
                                   float* __restrict__ out_rec) {
    const int lane   = threadIdx.x & 31;
    const int wid    = blockIdx.x * 4 + (threadIdx.x >> 5);   // 0..255
    const int mtile  = wid >> 6;                              // 0..3
    const int ntile  = wid & 63;                              // 0..63
    const int m0     = mtile * 16;
    const int n0     = ntile * 16;
    const int lane16 = lane & 15;
    const int khalf  = lane >> 4;

    const float* xrow = X  + (size_t)(m0 + lane16) * ISIZE;
    const float* wrow = W0 + (size_t)(n0 + lane16) * ISIZE;

    v8f c = {};

#pragma unroll 4
    for (int k0 = 0; k0 < ISIZE; k0 += 32) {
        const int kb = k0 + 8 * khalf;

        const float4 x0 = *(const float4*)(xrow + kb);
        const float4 x1 = *(const float4*)(xrow + kb + 4);
        const float4 x2 = *(const float4*)(xrow + kb + 16);
        const float4 x3 = *(const float4*)(xrow + kb + 20);
        const float4 y0 = *(const float4*)(wrow + kb);
        const float4 y1 = *(const float4*)(wrow + kb + 4);
        const float4 y2 = *(const float4*)(wrow + kb + 16);
        const float4 y3 = *(const float4*)(wrow + kb + 20);

        v16h a, bm;
        a[0]  = (_Float16)x0.x; a[1]  = (_Float16)x0.y; a[2]  = (_Float16)x0.z; a[3]  = (_Float16)x0.w;
        a[4]  = (_Float16)x1.x; a[5]  = (_Float16)x1.y; a[6]  = (_Float16)x1.z; a[7]  = (_Float16)x1.w;
        a[8]  = (_Float16)x2.x; a[9]  = (_Float16)x2.y; a[10] = (_Float16)x2.z; a[11] = (_Float16)x2.w;
        a[12] = (_Float16)x3.x; a[13] = (_Float16)x3.y; a[14] = (_Float16)x3.z; a[15] = (_Float16)x3.w;
        bm[0]  = (_Float16)y0.x; bm[1]  = (_Float16)y0.y; bm[2]  = (_Float16)y0.z; bm[3]  = (_Float16)y0.w;
        bm[4]  = (_Float16)y1.x; bm[5]  = (_Float16)y1.y; bm[6]  = (_Float16)y1.z; bm[7]  = (_Float16)y1.w;
        bm[8]  = (_Float16)y2.x; bm[9]  = (_Float16)y2.y; bm[10] = (_Float16)y2.z; bm[11] = (_Float16)y2.w;
        bm[12] = (_Float16)y3.x; bm[13] = (_Float16)y3.y; bm[14] = (_Float16)y3.z; bm[15] = (_Float16)y3.w;

        c = __builtin_amdgcn_wmma_f32_16x16x32_f16(
            /*neg_a=*/false, a, /*neg_b=*/false, bm,
            /*c_mod=*/(short)0, c, /*reuse_a=*/false, /*reuse_b=*/false);
    }

    const int n = n0 + lane16;
    const float bias = b0[n];
#pragma unroll
    for (int r = 0; r < 8; ++r) {
        const int brow = m0 + r + 8 * khalf;
        const float ri = partial[(size_t)0 * BS * REC + (size_t)brow * REC + n]
                       + partial[(size_t)1 * BS * REC + (size_t)brow * REC + n]
                       + partial[(size_t)2 * BS * REC + (size_t)brow * REC + n]
                       + partial[(size_t)3 * BS * REC + (size_t)brow * REC + n];
        out_rec[(size_t)brow * REC + n] = tanhf(c[r] + bias + ri);
    }
}

// ---------------------------------------------------------------------------
// Kernel 3: logits -> softmax (act_dis) and myeta = tanh(h @ h2mod_w + b).
// One block per batch row; LDS tree reductions.
// ---------------------------------------------------------------------------
__global__ void head_kernel(const float* __restrict__ hact,
                            const float* __restrict__ W1,
                            const float* __restrict__ b1,
                            const float* __restrict__ h2w,
                            const float* __restrict__ h2b,
                            float* __restrict__ act_out,
                            float* __restrict__ myeta) {
    __shared__ float red[256];
    __shared__ float logits_s[NACT];

    const int b = blockIdx.x;
    const int t = threadIdx.x;
    const float* h = hact + (size_t)b * REC;

    float part[NACT];
#pragma unroll
    for (int a = 0; a < NACT; ++a) part[a] = 0.f;
    float pm = 0.f;

    for (int j = t; j < REC; j += 256) {
        const float hv = h[j];
#pragma unroll
        for (int a = 0; a < NACT; ++a) part[a] += hv * W1[(size_t)a * REC + j];
        pm += hv * h2w[j];
    }

    for (int a = 0; a < NACT; ++a) {
        red[t] = part[a];
        __syncthreads();
        for (int s = 128; s > 0; s >>= 1) {
            if (t < s) red[t] += red[t + s];
            __syncthreads();
        }
        if (t == 0) logits_s[a] = red[0] + b1[a];
        __syncthreads();
    }

    red[t] = pm;
    __syncthreads();
    for (int s = 128; s > 0; s >>= 1) {
        if (t < s) red[t] += red[t + s];
        __syncthreads();
    }
    if (t == 0) myeta[b] = tanhf(red[0] + h2b[0]);
    __syncthreads();

    if (t == 0) {
        float mx = logits_s[0];
#pragma unroll
        for (int a = 1; a < NACT; ++a) mx = fmaxf(mx, logits_s[a]);
        float sum = 0.f;
        float e[NACT];
#pragma unroll
        for (int a = 0; a < NACT; ++a) { e[a] = __expf(logits_s[a] - mx); sum += e[a]; }
        const float inv = 1.f / sum;
#pragma unroll
        for (int a = 0; a < NACT; ++a) act_out[b * NACT + a] = e[a] * inv;
    }
}

// ---------------------------------------------------------------------------
// Kernel 4: hebb_out[b,i,j] = clip(hebb + (myeta[b]*mfw[j]+mfb[j]) * rv[b,i] * h[b,j]).
// grid = BS*16 (16 i-chunks per batch), 256 threads, float4 per thread over j.
// q[j] = eta[j]*h[j] is loop-invariant per thread. 268 MB read + 268 MB write.
// Unroll 4 keeps 4 b128 loads in flight per thread; prefetch hints four rows ahead.
// ---------------------------------------------------------------------------
__global__ void hebb_update_kernel(const float* __restrict__ hebb,
                                   const float* __restrict__ rv,
                                   const float* __restrict__ hact,
                                   const float* __restrict__ myeta,
                                   const float* __restrict__ mfw,
                                   const float* __restrict__ mfb,
                                   float* __restrict__ out_hebb) {
    const int b     = blockIdx.x >> 4;
    const int chunk = blockIdx.x & 15;
    const int j0    = threadIdx.x * 4;

    const float me = myeta[b];
    const float4 w4 = *(const float4*)(mfw + j0);
    const float4 b4 = *(const float4*)(mfb + j0);
    const float4 h4 = *(const float4*)(hact + (size_t)b * REC + j0);

    const float qx = (me * w4.x + b4.x) * h4.x;
    const float qy = (me * w4.y + b4.y) * h4.y;
    const float qz = (me * w4.z + b4.z) * h4.z;
    const float qw = (me * w4.w + b4.w) * h4.w;

    const float* rvb = rv + (size_t)b * REC;
    const size_t base = (size_t)b * REC * REC;

    const int i_begin = chunk * (REC / 16);
    const int i_end   = i_begin + (REC / 16);
#pragma unroll 4
    for (int i = i_begin; i < i_end; ++i) {
        const size_t off = base + (size_t)i * REC + j0;
        __builtin_prefetch(hebb + off + 4 * REC, 0, 0);   // global_prefetch_b8, 4 rows ahead
        const float s = rvb[i];
        const float4 hb = *(const float4*)(hebb + off);
        float4 o;
        o.x = fminf(fmaxf(hb.x + qx * s, -CLIPVAL), CLIPVAL);
        o.y = fminf(fmaxf(hb.y + qy * s, -CLIPVAL), CLIPVAL);
        o.z = fminf(fmaxf(hb.z + qz * s, -CLIPVAL), CLIPVAL);
        o.w = fminf(fmaxf(hb.w + qw * s, -CLIPVAL), CLIPVAL);
        *(float4*)(out_hebb + off) = o;
    }
}

// ---------------------------------------------------------------------------
extern "C" void kernel_launch(void* const* d_in, const int* in_sizes, int n_in,
                              void* d_out, int out_size, void* d_ws, size_t ws_size,
                              hipStream_t stream) {
    const float* X     = (const float*)d_in[0];   // [BS, ISIZE]
    const float* rv    = (const float*)d_in[1];   // [BS, REC]
    const float* hebb  = (const float*)d_in[2];   // [BS, REC, REC]
    const float* w     = (const float*)d_in[3];   // [REC, REC]
    const float* alpha = (const float*)d_in[4];   // [REC, REC]
    const float* W0    = (const float*)d_in[5];   // [REC, ISIZE]
    const float* b0    = (const float*)d_in[6];   // [REC]
    const float* W1    = (const float*)d_in[7];   // [NACT, REC]
    const float* b1    = (const float*)d_in[8];   // [NACT]
    const float* h2w   = (const float*)d_in[9];   // [1, REC]
    const float* h2b   = (const float*)d_in[10];  // [1]
    const float* mfw   = (const float*)d_in[11];  // [REC, 1]
    const float* mfb   = (const float*)d_in[12];  // [REC]

    float* out_act  = (float*)d_out;              // [BS, NACT]
    float* out_rec  = out_act + BS * NACT;        // [BS, REC]
    float* out_hebb = out_rec + BS * REC;         // [BS, REC, REC]

    float* partial = (float*)d_ws;                // [4, BS, REC]  (1 MB)
    float* myeta   = partial + 4 * BS * REC;      // [BS]

    // Pass 1: stream hebb once, reduce plastic recurrent input (partials).
    plastic_reduce_kernel<<<BS * 4, 256, 0, stream>>>(rv, hebb, w, alpha, partial);

    // Pass 2: WMMA GEMM + bias + rec_in + tanh -> rec_layer_values.
    wmma_hactiv_kernel<<<64, 128, 0, stream>>>(X, W0, b0, partial, out_rec);

    // Pass 3: softmax head + neuromodulator scalar.
    head_kernel<<<BS, 256, 0, stream>>>(out_rec, W1, b1, h2w, h2b, out_act, myeta);

    // Pass 4: stream hebb again, fused eta*deltahebb + clip -> hebb_return.
    hebb_update_kernel<<<BS * 16, 256, 0, stream>>>(hebb, rv, out_rec, myeta,
                                                    mfw, mfb, out_hebb);
}
